// SelfAttention_3375844294716
// MI455X (gfx1250) — matrix operand order repaired and checked by reading, not compiled, not run on hardware
//
#include <hip/hip_runtime.h>
#include <stdint.h>

// ---------------------------------------------------------------------------
// Self-attention block (GroupNorm -> QKV -> softmax(QK^T/sqrt(C))V -> proj + x)
// B=4, C=256, H=W=64, N=4096. BF16 WMMA (v_wmma_f32_16x16x32_bf16), f32 accum.
// Flash-attention style: scores never hit HBM. K-block staging uses the
// CDNA5 Tensor Data Mover (tensor_load_to_lds + s_wait_tensorcnt).
// ---------------------------------------------------------------------------

typedef __attribute__((ext_vector_type(16))) __bf16 v16bf;
typedef __attribute__((ext_vector_type(8)))  float  v8f;
typedef __attribute__((ext_vector_type(4)))  unsigned v4u;
typedef __attribute__((ext_vector_type(8)))  int      v8i;
typedef __attribute__((ext_vector_type(4)))  int      v4i;

#define B_   4
#define C_   256
#define N_   4096
#define C3_  768
#define SCALE_ 0.0625f   // C^-0.5 = 1/16

static __device__ __forceinline__ unsigned short f2bf(float f) {
  union { float f; unsigned u; } v; v.f = f;
  unsigned r = v.u + 0x7FFFu + ((v.u >> 16) & 1u);   // round-to-nearest-even
  return (unsigned short)(r >> 16);
}

union FragA { v16bf v; unsigned u[8]; };

// 16-bit A-matrix 16x32 layout (contiguous K, matching B): lane L -> row M=L&15;
// VGPR j: K = 2j,2j+1 (+16 if L>=16).
static __device__ __forceinline__ v16bf load_afrag(const unsigned short* rowptr,
                                                   int cbase, int half) {
  FragA f;
#pragma unroll
  for (int j = 0; j < 8; ++j) {
    int c0 = cbase + 2 * j + 16 * half;
    f.u[j] = *reinterpret_cast<const unsigned*>(rowptr + c0);  // packed K pair
  }
  return f.v;
}

static __device__ __forceinline__ v8f wmma_bf16(v16bf a, v16bf b, v8f c) {
  return __builtin_amdgcn_wmma_f32_16x16x32_bf16(false, a, false, b,
                                                 (short)0, c, false, false);
}

// ---------------------------------------------------------------------------
// TDM: async 2D tile load Global -> LDS.
// Tile: rows x cols bf16 elements, row stride `stride` elements.
// D# per CDNA5 ISA ch.8: group0 {count, lds_addr, global_addr, type=2},
// group1 {data_size=2B, tensor dims, tile dims, dim0 stride}.
// ---------------------------------------------------------------------------
#if __has_builtin(__builtin_amdgcn_tensor_load_to_lds)
#define HAVE_TDM 1
static __device__ __forceinline__ void tdm_load_2d(const void* gptr,
                                                   unsigned lds_off,
                                                   int cols, int rows, int stride) {
  unsigned long long ga = (unsigned long long)(uintptr_t)gptr;
  v4u g0;
  g0[0] = 1u;                                   // count=1 (valid descriptor)
  g0[1] = lds_off;                              // LDS byte address
  g0[2] = (unsigned)(ga & 0xFFFFFFFFull);       // global_addr[31:0]
  g0[3] = (unsigned)((ga >> 32) & 0x1FFFFFFull) | (2u << 30);  // [56:32] | type=2
  v8i g1;
  g1[0] = (int)(1u << 16);                      // data_size=1 -> 2 bytes (bf16)
  g1[1] = (int)((unsigned)cols << 16);          // tensor_dim0 = cols
  g1[2] = (int)((unsigned)rows << 16);          // tensor_dim1 = rows
  g1[3] = (int)((unsigned)cols << 16);          // tile_dim0 = cols
  g1[4] = rows;                                 // tile_dim1 = rows (tile_dim2=0)
  g1[5] = stride;                               // tensor_dim0_stride
  g1[6] = 0;
  g1[7] = 0;
  v4i z = {};
#if defined(__clang_major__) && (__clang_major__ >= 23)
  v8i z8 = {};
  __builtin_amdgcn_tensor_load_to_lds(g0, g1, z, z, z8, 0);
#else
  __builtin_amdgcn_tensor_load_to_lds(g0, g1, z, z, 0);
#endif
}
#else
#define HAVE_TDM 0
#endif

// ---------------------------------------------------------------------------
// Kernel 1: f32 -> bf16 weight conversion
// ---------------------------------------------------------------------------
__global__ void prep_weights(const float* __restrict__ qw,
                             const float* __restrict__ ow,
                             unsigned short* __restrict__ wq_bf,
                             unsigned short* __restrict__ wo_bf) {
  int i = blockIdx.x * 256 + threadIdx.x;
  if (i < C3_ * C_) wq_bf[i] = f2bf(qw[i]);
  if (i < C_ * C_)  wo_bf[i] = f2bf(ow[i]);
}

// ---------------------------------------------------------------------------
// Kernel 2: GroupNorm, writes xn transposed to [B, N, C] bf16
// ---------------------------------------------------------------------------
__global__ void __launch_bounds__(256)
groupnorm_kernel(const float* __restrict__ x, const float* __restrict__ w,
                 const float* __restrict__ bias, unsigned short* __restrict__ xn) {
  int b = blockIdx.x >> 5;
  int g = blockIdx.x & 31;
  int tid = threadIdx.x;
  const float* xg = x + ((size_t)(b * C_ + g * 8)) * N_;

  float s = 0.f, s2 = 0.f;
  for (int i = tid; i < 8 * N_; i += 256) {
    float v = xg[i];
    s += v; s2 += v * v;
  }
  __shared__ float rs[256], rq[256];
  rs[tid] = s; rq[tid] = s2;
  __syncthreads();
  for (int off = 128; off > 0; off >>= 1) {
    if (tid < off) { rs[tid] += rs[tid + off]; rq[tid] += rq[tid + off]; }
    __syncthreads();
  }
  const float inv = 1.f / (8.f * N_);
  float mean = rs[0] * inv;
  float var  = rq[0] * inv - mean * mean;
  float rstd = rsqrtf(var + 1e-5f);

  for (int i = tid; i < 8 * N_; i += 256) {
    int cl = i >> 12;
    int n  = i & (N_ - 1);
    int c  = g * 8 + cl;
    float v = (xg[(size_t)cl * N_ + n] - mean) * rstd * w[c] + bias[c];
    xn[((size_t)(b * N_ + n)) * C_ + c] = f2bf(v);
  }
}

// ---------------------------------------------------------------------------
// Kernel 3: QKV projection. The 1/sqrt(C) attention scale is folded into the
// Q third of the output (exact: softmax uses (xW+b)/16 . k).
// ---------------------------------------------------------------------------
__global__ void __launch_bounds__(256)
qkv_gemm(const unsigned short* __restrict__ xn,
         const unsigned short* __restrict__ wq_bf,
         const float* __restrict__ qkv_bias,
         unsigned short* __restrict__ qkvT) {
  int wave = threadIdx.x >> 5, lane = threadIdx.x & 31;
  int half = lane >> 4, ln = lane & 15;
  int b  = blockIdx.z;
  int n0 = blockIdx.x * 128 + wave * 16;
  int o0 = blockIdx.y * 128;

  v8f acc[8] = {};
  const unsigned short* arow = xn + ((size_t)(b * N_ + n0 + ln)) * C_;

#pragma unroll
  for (int kt = 0; kt < 8; ++kt) {
    v16bf a = load_afrag(arow, kt * 32, half);
#pragma unroll
    for (int ot = 0; ot < 8; ++ot) {
      int o = o0 + ot * 16 + ln;
      v16bf bf = *reinterpret_cast<const v16bf*>(
          wq_bf + (size_t)o * C_ + kt * 32 + 16 * half);
      acc[ot] = wmma_bf16(a, bf, acc[ot]);
    }
  }
#pragma unroll
  for (int ot = 0; ot < 8; ++ot) {
    int o = o0 + ot * 16 + ln;
    float bi  = qkv_bias[o];
    float mul = (o < C_) ? SCALE_ : 1.f;    // pre-scale Q rows
#pragma unroll
    for (int r = 0; r < 8; ++r) {
      int n = n0 + r + 8 * half;
      qkvT[((size_t)(b * N_ + n)) * C3_ + o] = f2bf((acc[ot][r] + bi) * mul);
    }
  }
}

// ---------------------------------------------------------------------------
// Kernel 4: flash attention. grid (N/128, B); 8 waves, wave -> 16 query rows.
// K block staged by TDM (wave 0), V block manually transposed in LDS.
// ---------------------------------------------------------------------------
__global__ void __launch_bounds__(256)
attention_kernel(const unsigned short* __restrict__ qkvT,
                 unsigned short* __restrict__ attn) {
  __shared__ __align__(16) unsigned short lds_k[32 * 256];   // [key][c] 16KB
  __shared__ __align__(16) unsigned short lds_v[256 * 32];   // [c][key] 16KB
  __shared__ __align__(16) unsigned short lds_p[8][512];     // per-wave P 8KB

  int tid = threadIdx.x;
  int wave = tid >> 5, lane = tid & 31;
  int half = lane >> 4, ln = lane & 15;
  int b  = blockIdx.y;
  int q0 = blockIdx.x * 128 + wave * 16;
  const unsigned short* base = qkvT + (size_t)b * N_ * C3_;

  v16bf qf[8];
  const unsigned short* qrow = base + (size_t)(q0 + ln) * C3_;
#pragma unroll
  for (int kt = 0; kt < 8; ++kt) qf[kt] = load_afrag(qrow, kt * 32, half);

  v8f acc[16] = {};
  float mrow[8], lrow[8];
#pragma unroll
  for (int r = 0; r < 8; ++r) { mrow[r] = -1e30f; lrow[r] = 0.f; }

  for (int kb = 0; kb < N_ / 32; ++kb) {
    __syncthreads();
    const unsigned short* kb_ptr = base + (size_t)(kb * 32) * C3_ + C_;
    const unsigned short* vb_ptr = base + (size_t)(kb * 32) * C3_ + 2 * C_;

#if HAVE_TDM
    if (wave == 0) {                     // one TDM descriptor per workgroup
      tdm_load_2d(kb_ptr, (unsigned)(uintptr_t)&lds_k[0], C_, 32, C3_);
    }
#else
#pragma unroll
    for (int it = 0; it < 4; ++it) {
      int idx8 = tid + it * 256;
      int key  = idx8 >> 5;
      int coff = (idx8 & 31) * 8;
      *reinterpret_cast<uint4*>(&lds_k[key * 256 + coff]) =
          *reinterpret_cast<const uint4*>(kb_ptr + (size_t)key * C3_ + coff);
    }
#endif
#pragma unroll
    for (int it = 0; it < 16; ++it) {    // V block transposed: lds_v[c][key]
      int idx2 = tid + it * 256;
      int key  = idx2 >> 7;
      int c    = (idx2 & 127) * 2;
      unsigned pv = *reinterpret_cast<const unsigned*>(vb_ptr + (size_t)key * C3_ + c);
      lds_v[c * 32 + key]       = (unsigned short)(pv & 0xFFFFu);
      lds_v[(c + 1) * 32 + key] = (unsigned short)(pv >> 16);
    }
    if (kb + 1 < N_ / 32)                // global_prefetch_b8 of next V block
      __builtin_prefetch(vb_ptr + (size_t)32 * C3_ + (tid & 31) * 64, 0, 0);
#if HAVE_TDM
    if (wave == 0) __builtin_amdgcn_s_wait_tensorcnt(0);
#endif
    __syncthreads();

    // S = Q * K^T : two 16x16 D tiles (32 keys); scale pre-folded into Q
    v8f s0 = {}, s1 = {};
#pragma unroll
    for (int kt = 0; kt < 8; ++kt) {
      v16bf b0 = *reinterpret_cast<const v16bf*>(&lds_k[ln * 256 + kt * 32 + 16 * half]);
      v16bf b1 = *reinterpret_cast<const v16bf*>(&lds_k[(16 + ln) * 256 + kt * 32 + 16 * half]);
      s0 = wmma_bf16(qf[kt], b0, s0);
      s1 = wmma_bf16(qf[kt], b1, s1);
    }

    // online softmax: row r+8*half lives in one 16-lane half -> xor reduce
    float p0[8], p1[8], sc[8];
#pragma unroll
    for (int r = 0; r < 8; ++r) {
      float a0 = s0[r], a1 = s1[r];
      float rmax = fmaxf(a0, a1);
      rmax = fmaxf(rmax, __shfl_xor(rmax, 1));
      rmax = fmaxf(rmax, __shfl_xor(rmax, 2));
      rmax = fmaxf(rmax, __shfl_xor(rmax, 4));
      rmax = fmaxf(rmax, __shfl_xor(rmax, 8));
      float newm = fmaxf(mrow[r], rmax);
      sc[r] = __expf(mrow[r] - newm);
      mrow[r] = newm;
      p0[r] = __expf(a0 - newm);
      p1[r] = __expf(a1 - newm);
      float rsum = p0[r] + p1[r];
      rsum += __shfl_xor(rsum, 1);
      rsum += __shfl_xor(rsum, 2);
      rsum += __shfl_xor(rsum, 4);
      rsum += __shfl_xor(rsum, 8);
      lrow[r] = lrow[r] * sc[r] + rsum;
    }
#pragma unroll
    for (int ct = 0; ct < 16; ++ct)
#pragma unroll
      for (int r = 0; r < 8; ++r) acc[ct][r] *= sc[r];

    // re-layout P (D tiles -> A fragment) through per-wave LDS tile
#pragma unroll
    for (int r = 0; r < 8; ++r) {
      int row = r + 8 * half;
      lds_p[wave][row * 32 + ln]      = f2bf(p0[r]);
      lds_p[wave][row * 32 + 16 + ln] = f2bf(p1[r]);
    }
    v16bf pa = load_afrag(&lds_p[wave][(size_t)ln * 32], 0, half);

    // O += P * V
#pragma unroll
    for (int ct = 0; ct < 16; ++ct) {
      v16bf vb = *reinterpret_cast<const v16bf*>(&lds_v[(ct * 16 + ln) * 32 + 16 * half]);
      acc[ct] = wmma_bf16(pa, vb, acc[ct]);
    }
  }

#pragma unroll
  for (int r = 0; r < 8; ++r) lrow[r] = 1.f / lrow[r];
  unsigned short* orow = attn + (size_t)b * N_ * C_;
#pragma unroll
  for (int ct = 0; ct < 16; ++ct)
#pragma unroll
    for (int r = 0; r < 8; ++r) {
      int n = q0 + r + 8 * half;
      orow[(size_t)n * C_ + ct * 16 + ln] = f2bf(acc[ct][r] * lrow[r]);
    }
}

// ---------------------------------------------------------------------------
// Kernel 5: out projection + bias + residual -> d_out [B, C, H, W] f32
// ---------------------------------------------------------------------------
__global__ void __launch_bounds__(256)
outproj_kernel(const unsigned short* __restrict__ attn,
               const unsigned short* __restrict__ wo_bf,
               const float* __restrict__ out_bias,
               const float* __restrict__ x, float* __restrict__ out) {
  int wave = threadIdx.x >> 5, lane = threadIdx.x & 31;
  int half = lane >> 4, ln = lane & 15;
  int b  = blockIdx.y;
  int n0 = blockIdx.x * 128 + wave * 16;

  v8f acc[16] = {};
  const unsigned short* arow = attn + ((size_t)(b * N_ + n0 + ln)) * C_;
#pragma unroll
  for (int kt = 0; kt < 8; ++kt) {
    v16bf a = load_afrag(arow, kt * 32, half);
#pragma unroll
    for (int ot = 0; ot < 16; ++ot) {
      v16bf bf = *reinterpret_cast<const v16bf*>(
          wo_bf + (size_t)(ot * 16 + ln) * C_ + kt * 32 + 16 * half);
      acc[ot] = wmma_bf16(a, bf, acc[ot]);
    }
  }
#pragma unroll
  for (int ot = 0; ot < 16; ++ot) {
    int o = ot * 16 + ln;
    float bi = out_bias[o];
#pragma unroll
    for (int r = 0; r < 8; ++r) {
      int n = n0 + r + 8 * half;
      size_t idx = ((size_t)(b * C_ + o)) * N_ + n;
      out[idx] = acc[ot][r] + bi + x[idx];
    }
  }
}

// ---------------------------------------------------------------------------
extern "C" void kernel_launch(void* const* d_in, const int* in_sizes, int n_in,
                              void* d_out, int out_size, void* d_ws, size_t ws_size,
                              hipStream_t stream) {
  const float* x      = (const float*)d_in[0];
  const float* gn_w   = (const float*)d_in[1];
  const float* gn_b   = (const float*)d_in[2];
  const float* qkv_w  = (const float*)d_in[3];
  const float* qkv_b  = (const float*)d_in[4];
  const float* out_w  = (const float*)d_in[5];
  const float* out_b  = (const float*)d_in[6];
  float* out = (float*)d_out;

  char* ws = (char*)d_ws;
  size_t off = 0;
  unsigned short* wq_bf = (unsigned short*)(ws + off); off += (size_t)C3_ * C_ * 2;
  unsigned short* wo_bf = (unsigned short*)(ws + off); off += (size_t)C_ * C_ * 2;
  unsigned short* xn    = (unsigned short*)(ws + off); off += (size_t)B_ * N_ * C_ * 2;
  unsigned short* qkvT  = (unsigned short*)(ws + off); off += (size_t)B_ * N_ * C3_ * 2;
  unsigned short* attnb = (unsigned short*)(ws + off); off += (size_t)B_ * N_ * C_ * 2;
  if (ws_size < off) return;   // ~41 MB needed

  prep_weights<<<dim3((C3_ * C_) / 256), 256, 0, stream>>>(qkv_w, out_w, wq_bf, wo_bf);
  groupnorm_kernel<<<dim3(B_ * 32), 256, 0, stream>>>(x, gn_w, gn_b, xn);
  qkv_gemm<<<dim3(N_ / 128, C3_ / 128, B_), 256, 0, stream>>>(xn, wq_bf, qkv_b, qkvT);
  attention_kernel<<<dim3(N_ / 128, B_), 256, 0, stream>>>(qkvT, attnb);
  outproj_kernel<<<dim3(N_ / 128, B_), 256, 0, stream>>>(attnb, wo_bf, out_b, x, out);
}
